// RingAttention_17918603559266
// MI455X (gfx1250) — matrix operand order repaired and checked by reading, not compiled, not run on hardware
//
#include <hip/hip_runtime.h>
#include <hip/hip_bf16.h>

// ---------------------------------------------------------------------------
// Flash attention B=1, S=4096, H=16, D=64 (fp32 in/out), bf16 WMMA core.
//
// Fast path (needs 16MB workspace):
//   1) conv_bf16: one-shot fp32->bf16 conversion of K (row-major per head) and
//      V (transposed per head) into d_ws. Each element converted once instead
//      of once per query-tile block (64x saving), V transpose out of hot loop.
//   2) flash_fast: double-buffered GLOBAL_LOAD_ASYNC_TO_LDS_B128 staging
//      (ASYNCcnt), 16 bf16 WMMAs per 64-key tile, DPP softmax reductions.
// Fallback path: fused load+convert kernel (previous round).
// ---------------------------------------------------------------------------

typedef __attribute__((ext_vector_type(16))) __bf16 v16bf;
typedef __attribute__((ext_vector_type(8)))  __bf16 v8bf;
typedef __attribute__((ext_vector_type(8)))  float  v8f;
typedef int v4i_vs __attribute__((vector_size(16)));  // matches builtin param

namespace cfg {
constexpr int S  = 4096;
constexpr int H  = 16;
constexpr int D  = 64;
constexpr int QT = 64;          // queries per block
constexpr int KT = 64;          // keys per tile
constexpr int NT = S / KT;      // 64 key tiles
constexpr int WAVES = QT / 16;  // 4 waves
// 1/sqrt(64) * log2(e): softmax exponentials become bare v_exp_f32.
constexpr float SCALE = 0.125f * 1.44269504088896340736f;
}

// ---------------------------------------------------------------------------
// helpers
// ---------------------------------------------------------------------------
__device__ __forceinline__ v16bf ld16(const __bf16* p) {
  v8bf lo = *(const v8bf*)(p);
  v8bf hi = *(const v8bf*)(p + 8);
  return __builtin_shufflevector(lo, hi,
      0, 1, 2, 3, 4, 5, 6, 7, 8, 9, 10, 11, 12, 13, 14, 15);
}

__device__ __forceinline__ v8f wmma_bf16(v16bf a, v16bf b, v8f c) {
  return __builtin_amdgcn_wmma_f32_16x16x32_bf16(
      false, a, false, b, (short)0, c, false, false);
}

template <int CTRL>
__device__ __forceinline__ float dpp_bcast_f32(float v) {
  const int t = __builtin_amdgcn_update_dpp(
      0, __float_as_int(v), CTRL, 0xF, 0xF, true);
  return __int_as_float(t);
}
__device__ __forceinline__ float red_max16(float v) {
  v = fmaxf(v, dpp_bcast_f32<0xB1>(v));
  v = fmaxf(v, dpp_bcast_f32<0x4E>(v));
  v = fmaxf(v, dpp_bcast_f32<0x141>(v));
  v = fmaxf(v, dpp_bcast_f32<0x140>(v));
  return v;
}
__device__ __forceinline__ float red_sum16(float v) {
  v += dpp_bcast_f32<0xB1>(v);
  v += dpp_bcast_f32<0x4E>(v);
  v += dpp_bcast_f32<0x141>(v);
  v += dpp_bcast_f32<0x140>(v);
  return v;
}

// Async global->LDS copy, 16B per lane, tracked by ASYNCcnt.
__device__ __forceinline__ void async_b128(const void* g, void* l) {
  __builtin_amdgcn_global_load_async_to_lds_b128(
      (__attribute__((address_space(1))) v4i_vs*)(g),
      (__attribute__((address_space(3))) v4i_vs*)(l),
      0, 0);
}
template <int N>
__device__ __forceinline__ void wait_async() {
  asm volatile("s_wait_asynccnt %0" :: "i"(N));
}

// Load Q rows into the two 16x32 bf16 A-fragments (scale folded in).
__device__ __forceinline__ void load_q_frags(const float* qbase, int half,
                                             v16bf aq[2]) {
  using namespace cfg;
#pragma unroll
  for (int s = 0; s < 2; ++s) {
    v16bf a;
#pragma unroll
    for (int c = 0; c < 2; ++c) {
      const int kb = s * 32 + half * 16 + c * 8;
      const float4 f0 = *(const float4*)(qbase + kb);
      const float4 f1 = *(const float4*)(qbase + kb + 4);
      a[c * 8 + 0] = (__bf16)(f0.x * SCALE);
      a[c * 8 + 1] = (__bf16)(f0.y * SCALE);
      a[c * 8 + 2] = (__bf16)(f0.z * SCALE);
      a[c * 8 + 3] = (__bf16)(f0.w * SCALE);
      a[c * 8 + 4] = (__bf16)(f1.x * SCALE);
      a[c * 8 + 5] = (__bf16)(f1.y * SCALE);
      a[c * 8 + 6] = (__bf16)(f1.z * SCALE);
      a[c * 8 + 7] = (__bf16)(f1.w * SCALE);
    }
    aq[s] = a;
  }
}

// ---------------------------------------------------------------------------
// Pre-pass: K -> Kb[h][s][d] (bf16), V -> Vt[h][d][s] (bf16, transposed).
// One block per (head, 64-key tile); transpose staged through LDS.
// ---------------------------------------------------------------------------
__global__ __launch_bounds__(128)
void conv_bf16(const float* __restrict__ K, const float* __restrict__ V,
               __bf16* __restrict__ Kb, __bf16* __restrict__ Vt) {
  using namespace cfg;
  const int h   = blockIdx.x % H;
  const int kt  = blockIdx.x / H;
  const int tid = threadIdx.x;
  const int kk  = tid >> 1;   // key in tile (0..63)
  const int seg = tid & 1;    // 32-dim half

  __shared__ __attribute__((aligned(16))) __bf16 sT[D * KT];  // [dim][key]

  const int kg = kt * KT + kk;
  const float* kbp = K + ((long)kg * H + h) * D + seg * 32;
  const float* vbp = V + ((long)kg * H + h) * D + seg * 32;

  __bf16 krow[32];
#pragma unroll
  for (int i = 0; i < 32; i += 4) {
    const float4 fk = *(const float4*)(kbp + i);
    krow[i + 0] = (__bf16)fk.x;  krow[i + 1] = (__bf16)fk.y;
    krow[i + 2] = (__bf16)fk.z;  krow[i + 3] = (__bf16)fk.w;
    const float4 fv = *(const float4*)(vbp + i);
    sT[(seg * 32 + i + 0) * KT + kk] = (__bf16)fv.x;
    sT[(seg * 32 + i + 1) * KT + kk] = (__bf16)fv.y;
    sT[(seg * 32 + i + 2) * KT + kk] = (__bf16)fv.z;
    sT[(seg * 32 + i + 3) * KT + kk] = (__bf16)fv.w;
  }
  // K out: contiguous 32 bf16 (64B) per thread.
  __bf16* ko = Kb + ((long)h * S + kg) * D + seg * 32;
#pragma unroll
  for (int i = 0; i < 32; i += 8) *(v8bf*)(ko + i) = *(const v8bf*)(krow + i);

  __syncthreads();
  // Vt out: thread t -> dim = t/2, 32 contiguous keys (64B).
  const int dim = tid >> 1;
  __bf16* vo = Vt + ((long)h * D + dim) * S + (long)kt * KT + seg * 32;
  const __bf16* vi = &sT[dim * KT + seg * 32];
#pragma unroll
  for (int i = 0; i < 32; i += 8) *(v8bf*)(vo + i) = *(const v8bf*)(vi + i);
}

// ---------------------------------------------------------------------------
// Fast path: async double-buffered staging from bf16 workspace.
// ---------------------------------------------------------------------------
__global__ __launch_bounds__(128)
void flash_fast(const float* __restrict__ Q,
                const __bf16* __restrict__ Kb,
                const __bf16* __restrict__ Vt,
                float* __restrict__ Out) {
  using namespace cfg;

  const int blk  = blockIdx.x;
  const int h    = blk % H;
  const int qt   = blk / H;
  const int tid  = threadIdx.x;
  const int wave = tid >> 5;
  const int lane = tid & 31;
  const int half = lane >> 4;
  const int l16  = lane & 15;

  __shared__ __attribute__((aligned(16))) __bf16 sKb[2][KT * D];      // [key][dim]
  __shared__ __attribute__((aligned(16))) __bf16 sVb[2][D * KT];      // [dim][key]
  __shared__ __attribute__((aligned(16))) __bf16 sP[WAVES][16 * KT];  // per-wave P

  const __bf16* Kh  = Kb + (long)h * S * D;   // [S][64] bf16, tiles contiguous
  const __bf16* Vth = Vt + (long)h * D * S;   // [64][S] bf16

  // Issue this wave's share (8 async b128 ops = 4KB) of tile kt into buf b.
  auto issue = [&](int kt, int b) {
    const char* gK = (const char*)Kh + (size_t)kt * (KT * D * 2);  // 8KB blob
    char* lK = (char*)&sKb[b][0];
#pragma unroll
    for (int i = 0; i < 4; ++i) {
      const int off = wave * 2048 + i * 512 + lane * 16;
      async_b128(gK + off, lK + off);
    }
    const char* gV = (const char*)Vth + (size_t)kt * (KT * 2);     // col offset
    char* lV = (char*)&sVb[b][0];
#pragma unroll
    for (int i = 0; i < 4; ++i) {
      const int row = (wave * 4 + i) * 4 + (lane >> 3);  // 0..63 dims
      const int col = (lane & 7) * 16;                   // 0..112 bytes
      async_b128(gV + (size_t)row * (S * 2) + col, lV + row * 128 + col);
    }
  };

  const int qrow = qt * QT + wave * 16 + l16;
  v16bf aq[2];
  load_q_frags(Q + ((long)qrow * H + h) * D, half, aq);

  float m[8], l[8];
  v8f o[4];
#pragma unroll
  for (int r = 0; r < 8; ++r) { m[r] = -3.0e38f; l[r] = 0.0f; }
#pragma unroll
  for (int g = 0; g < 4; ++g) o[g] = (v8f){0, 0, 0, 0, 0, 0, 0, 0};

  issue(0, 0);

  for (int kt = 0; kt < NT; ++kt) {
    const int cur = kt & 1;
    if (kt + 1 < NT) {
      issue(kt + 1, cur ^ 1);
      wait_async<8>();   // tile kt's 8 ops done; next tile's 8 may be in flight
    } else {
      wait_async<0>();
    }
    __syncthreads();

    const __bf16* sk = &sKb[cur][0];
    const __bf16* sv = &sVb[cur][0];

    // ---- QK^T: 4 C fragments of 16 keys each.
    v8f c[4];
#pragma unroll
    for (int g = 0; g < 4; ++g) {
      const v16bf b0 = ld16(&sk[(g * 16 + l16) * D + 0 * 32 + half * 16]);
      const v16bf b1 = ld16(&sk[(g * 16 + l16) * D + 1 * 32 + half * 16]);
      v8f acc = (v8f){0, 0, 0, 0, 0, 0, 0, 0};
      acc = wmma_bf16(aq[0], b0, acc);
      acc = wmma_bf16(aq[1], b1, acc);
      c[g] = acc;
    }

    // ---- Online softmax (log2 domain).
    float corr[8];
#pragma unroll
    for (int r = 0; r < 8; ++r) {
      float tmax = fmaxf(fmaxf(c[0][r], c[1][r]), fmaxf(c[2][r], c[3][r]));
      tmax = red_max16(tmax);
      const float mn = fmaxf(m[r], tmax);
      corr[r] = exp2f(m[r] - mn);
      m[r] = mn;
      float rs = 0.0f;
#pragma unroll
      for (int g = 0; g < 4; ++g) {
        c[g][r] = exp2f(c[g][r] - mn);
        rs += c[g][r];
      }
      l[r] = l[r] * corr[r] + red_sum16(rs);
    }
#pragma unroll
    for (int g = 0; g < 4; ++g)
#pragma unroll
      for (int r = 0; r < 8; ++r) o[g][r] *= corr[r];

    // ---- P: C layout -> A layout via per-wave LDS scratch.
    __bf16* pw = sP[wave];
#pragma unroll
    for (int r = 0; r < 8; ++r) {
      const int row = r + half * 8;
#pragma unroll
      for (int g = 0; g < 4; ++g) pw[row * KT + g * 16 + l16] = (__bf16)c[g][r];
    }
    v16bf ap[2];
#pragma unroll
    for (int s = 0; s < 2; ++s) {
      const v8bf p0 = *(const v8bf*)&pw[l16 * KT + s * 32 + half * 16];
      const v8bf p1 = *(const v8bf*)&pw[l16 * KT + s * 32 + half * 16 + 8];
      ap[s] = __builtin_shufflevector(p0, p1,
          0, 1, 2, 3, 4, 5, 6, 7, 8, 9, 10, 11, 12, 13, 14, 15);
    }

    // ---- O += P * V.
#pragma unroll
    for (int g = 0; g < 4; ++g) {
      const v16bf bv0 = ld16(&sv[(g * 16 + l16) * KT + 0 * 32 + half * 16]);
      const v16bf bv1 = ld16(&sv[(g * 16 + l16) * KT + 1 * 32 + half * 16]);
      o[g] = wmma_bf16(ap[0], bv0, o[g]);
      o[g] = wmma_bf16(ap[1], bv1, o[g]);
    }

    if (kt + 1 < NT) __syncthreads();  // readers done before buffer reuse
  }

#pragma unroll
  for (int r = 0; r < 8; ++r) {
    const float inv = 1.0f / l[r];
    const int row   = qt * QT + wave * 16 + r + half * 8;
    float* ob = Out + ((long)row * H + h) * D;
#pragma unroll
    for (int g = 0; g < 4; ++g) ob[g * 16 + l16] = o[g][r] * inv;
  }
}

// ---------------------------------------------------------------------------
// Fallback: fused load+convert kernel (no workspace needed).
// ---------------------------------------------------------------------------
__global__ __launch_bounds__(128)
void flash_fallback(const float* __restrict__ Q, const float* __restrict__ K,
                    const float* __restrict__ V, float* __restrict__ Out) {
  using namespace cfg;
  const int blk  = blockIdx.x;
  const int h    = blk % H;
  const int qt   = blk / H;
  const int tid  = threadIdx.x;
  const int wave = tid >> 5;
  const int lane = tid & 31;
  const int half = lane >> 4;
  const int l16  = lane & 15;

  __shared__ __attribute__((aligned(16))) __bf16 sK[KT * D];
  __shared__ __attribute__((aligned(16))) __bf16 sVt[D * KT];
  __shared__ __attribute__((aligned(16))) __bf16 sP[WAVES][16 * KT];

  const int qrow = qt * QT + wave * 16 + l16;
  v16bf aq[2];
  load_q_frags(Q + ((long)qrow * H + h) * D, half, aq);

  float m[8], l[8];
  v8f o[4];
#pragma unroll
  for (int r = 0; r < 8; ++r) { m[r] = -3.0e38f; l[r] = 0.0f; }
#pragma unroll
  for (int g = 0; g < 4; ++g) o[g] = (v8f){0, 0, 0, 0, 0, 0, 0, 0};

  const int kk  = tid >> 1;
  const int seg = tid & 1;

  for (int kt = 0; kt < NT; ++kt) {
    __syncthreads();
    const int kg = kt * KT + kk;
    const float* kb = K + ((long)kg * H + h) * D + seg * 32;
    const float* vb = V + ((long)kg * H + h) * D + seg * 32;
#pragma unroll
    for (int i = 0; i < 32; i += 4) {
      const float4 fk = *(const float4*)(kb + i);
      sK[kk * D + seg * 32 + i + 0] = (__bf16)fk.x;
      sK[kk * D + seg * 32 + i + 1] = (__bf16)fk.y;
      sK[kk * D + seg * 32 + i + 2] = (__bf16)fk.z;
      sK[kk * D + seg * 32 + i + 3] = (__bf16)fk.w;
      const float4 fv = *(const float4*)(vb + i);
      sVt[(seg * 32 + i + 0) * KT + kk] = (__bf16)fv.x;
      sVt[(seg * 32 + i + 1) * KT + kk] = (__bf16)fv.y;
      sVt[(seg * 32 + i + 2) * KT + kk] = (__bf16)fv.z;
      sVt[(seg * 32 + i + 3) * KT + kk] = (__bf16)fv.w;
    }
    __syncthreads();

    v8f c[4];
#pragma unroll
    for (int g = 0; g < 4; ++g) {
      const v16bf b0 = ld16(&sK[(g * 16 + l16) * D + 0 * 32 + half * 16]);
      const v16bf b1 = ld16(&sK[(g * 16 + l16) * D + 1 * 32 + half * 16]);
      v8f acc = (v8f){0, 0, 0, 0, 0, 0, 0, 0};
      acc = wmma_bf16(aq[0], b0, acc);
      acc = wmma_bf16(aq[1], b1, acc);
      c[g] = acc;
    }

    float corr[8];
#pragma unroll
    for (int r = 0; r < 8; ++r) {
      float tmax = fmaxf(fmaxf(c[0][r], c[1][r]), fmaxf(c[2][r], c[3][r]));
      tmax = red_max16(tmax);
      const float mn = fmaxf(m[r], tmax);
      corr[r] = exp2f(m[r] - mn);
      m[r] = mn;
      float rs = 0.0f;
#pragma unroll
      for (int g = 0; g < 4; ++g) {
        c[g][r] = exp2f(c[g][r] - mn);
        rs += c[g][r];
      }
      l[r] = l[r] * corr[r] + red_sum16(rs);
    }
#pragma unroll
    for (int g = 0; g < 4; ++g)
#pragma unroll
      for (int r = 0; r < 8; ++r) o[g][r] *= corr[r];

    __bf16* pw = sP[wave];
#pragma unroll
    for (int r = 0; r < 8; ++r) {
      const int row = r + half * 8;
#pragma unroll
      for (int g = 0; g < 4; ++g) pw[row * KT + g * 16 + l16] = (__bf16)c[g][r];
    }
    v16bf ap[2];
#pragma unroll
    for (int s = 0; s < 2; ++s) {
      const v8bf p0 = *(const v8bf*)&pw[l16 * KT + s * 32 + half * 16];
      const v8bf p1 = *(const v8bf*)&pw[l16 * KT + s * 32 + half * 16 + 8];
      ap[s] = __builtin_shufflevector(p0, p1,
          0, 1, 2, 3, 4, 5, 6, 7, 8, 9, 10, 11, 12, 13, 14, 15);
    }
#pragma unroll
    for (int g = 0; g < 4; ++g) {
      const v16bf bv0 = ld16(&sVt[(g * 16 + l16) * KT + 0 * 32 + half * 16]);
      const v16bf bv1 = ld16(&sVt[(g * 16 + l16) * KT + 1 * 32 + half * 16]);
      o[g] = wmma_bf16(ap[0], bv0, o[g]);
      o[g] = wmma_bf16(ap[1], bv1, o[g]);
    }
  }

#pragma unroll
  for (int r = 0; r < 8; ++r) {
    const float inv = 1.0f / l[r];
    const int row   = qt * QT + wave * 16 + r + half * 8;
    float* ob = Out + ((long)row * H + h) * D;
#pragma unroll
    for (int g = 0; g < 4; ++g) ob[g * 16 + l16] = o[g][r] * inv;
  }
}

// ---------------------------------------------------------------------------
extern "C" void kernel_launch(void* const* d_in, const int* in_sizes, int n_in,
                              void* d_out, int out_size, void* d_ws, size_t ws_size,
                              hipStream_t stream) {
  (void)in_sizes; (void)n_in; (void)out_size;
  using namespace cfg;
  const float* Q = (const float*)d_in[0];
  const float* K = (const float*)d_in[1];
  const float* V = (const float*)d_in[2];
  float* O = (float*)d_out;

  const dim3 grid(H * (S / QT));  // 1024 blocks
  const dim3 block(32 * WAVES);   // 128 threads

  const size_t one = (size_t)S * H * D * sizeof(__bf16);  // 8 MB
  if (ws_size >= 2 * one) {
    __bf16* Kb = (__bf16*)d_ws;
    __bf16* Vt = (__bf16*)((char*)d_ws + one);
    conv_bf16<<<grid, block, 0, stream>>>(K, V, Kb, Vt);
    flash_fast<<<grid, block, 0, stream>>>(Q, Kb, Vt, O);
  } else {
    flash_fallback<<<grid, block, 0, stream>>>(Q, K, V, O);
  }
}
